// SANet_34093450396107
// MI455X (gfx1250) — compile-verified, matrix operand order checked by reading, not run on hardware
//
#include <hip/hip_runtime.h>

// ---------------------------------------------------------------------------
// CDNA5 (gfx1250) SANet forward.
// Every conv = implicit GEMM on v_wmma_f32_16x16x32_bf16:
//   - weights pre-packed (per layer, per launch) to zero-padded bf16 [Cout][Kp]
//   - 256 threads / 8 wave32s per block, tile 32(M) x 128(N) x 64(K)
//   - each wave computes a 16x32 C strip -> 4 WMMAs per K-stage
//   - A-tile staged via GLOBAL_LOAD_ASYNC_TO_LDS_B128 when available
//   - BN-fold + SiLU/ReLU + residual fused in epilogue (f32 in HBM)
// ---------------------------------------------------------------------------

#define ACT_NONE 0
#define ACT_SILU 1
#define ACT_RELU 2

typedef __attribute__((ext_vector_type(16))) __bf16         v16bf;
typedef __attribute__((ext_vector_type(8)))  float          v8f;
typedef __attribute__((ext_vector_type(8)))  unsigned short ushort8;

union FragBF { ushort8 h[2]; v16bf v; };

#if defined(__has_builtin)
#if __has_builtin(__builtin_amdgcn_global_load_async_to_lds_b128) && \
    __has_builtin(__builtin_amdgcn_s_wait_asynccnt)
#define USE_ASYNC_LDS 1
#endif
#endif

#ifdef USE_ASYNC_LDS
typedef int v4i_ __attribute__((vector_size(16)));
typedef __attribute__((address_space(1))) v4i_* gv4p;   // global int4*
typedef __attribute__((address_space(3))) v4i_* lv4p;   // LDS    int4*
#endif

__device__ __forceinline__ unsigned short f2bf(float f) {
  unsigned int u = __float_as_uint(f);
  u += 0x7FFFu + ((u >> 16) & 1u);      // round-to-nearest-even
  return (unsigned short)(u >> 16);
}

// pack f32 (Cout, Kg) weights -> bf16 (Cout, Kp) zero-padded
__global__ void k_packw(const float* __restrict__ w, unsigned short* __restrict__ wp,
                        int Cout, int Kg, int Kp) {
  size_t n = (size_t)Cout * Kp;
  size_t i = (size_t)blockIdx.x * blockDim.x + threadIdx.x;
  if (i >= n) return;
  int kp = (int)(i % Kp);
  int oc = (int)(i / Kp);
  wp[i] = (kp < Kg) ? f2bf(w[(size_t)oc * Kg + kp]) : (unsigned short)0;
}

// ---------------------------------------------------------------------------
// conv_wmma: M = C_out/group, N = B*Ho*Wo, K = Cin_g*ks*ks (padded to 64).
// blockIdx = (N/128 tiles, M/32 tiles, group).
// ---------------------------------------------------------------------------
__global__ __launch_bounds__(256) void conv_wmma(
    const float* __restrict__ x, const unsigned short* __restrict__ wp,
    const float* __restrict__ bn_g, const float* __restrict__ bn_b,
    const float* __restrict__ bn_m, const float* __restrict__ bn_v,
    const float* __restrict__ res, float* __restrict__ y,
    int B, int Cin, int H, int W, int Cout, int Ho, int Wo,
    int ks, int stride, int pad, int dil, int groups, int Kg, int Kp, int act)
{
  __shared__ unsigned short As[32][64];    // 4 KB  (weights, bf16)
  __shared__ unsigned short Bs[128][64];   // 16 KB (im2col,  bf16)

  const int tid  = threadIdx.x;
  const int lane = tid & 31;
  const int wave = tid >> 5;
  const int wm   = wave >> 2;              // 0..1 : M sub-tile
  const int wn   = wave & 3;               // 0..3 : N sub-tile (32 wide)

  const int cing  = Cin  / groups;
  const int coutg = Cout / groups;
  const int khw   = ks * ks;
  const int gidx  = blockIdx.z;
  const int mBlk  = blockIdx.y * 32;       // within group
  const int nBlk  = blockIdx.x * 128;
  const int Ntot  = B * Ho * Wo;
  const int HoWo  = Ho * Wo;
  const int HW    = H * W;

  // ---- per-thread A staging setup: 8 contiguous k's of one weight row ----
  const int aIdx = tid * 8;                // 2048 = 32*64 elements
  const int amL  = aIdx >> 6;
  const int akk  = aIdx & 63;
  const bool aValid = (mBlk + amL) < coutg;
  const unsigned short* aRow =
      wp + (size_t)(gidx * coutg + mBlk + amL) * Kp + akk;

  // ---- per-thread B staging setup: one n row, 32 contiguous k's ----
  const int bRow = tid >> 1;               // 0..127
  const int bKK  = (tid & 1) * 32;
  const int nIdx = nBlk + bRow;
  const bool nValid = nIdx < Ntot;
  int hib = 0, wib = 0;
  const float* xb = x;
  {
    int b  = nIdx / HoWo, hw = nIdx - b * HoWo;
    int ho = hw / Wo,     wo = hw - ho * Wo;
    hib = ho * stride - pad;
    wib = wo * stride - pad;
    xb  = x + ((size_t)b * Cin + gidx * cing) * HW;
  }

  v8f acc[2] = {};

  for (int k0 = 0; k0 < Kg; k0 += 64) {
    // ---- stage A: 32x64 packed-bf16 weights ----
    if (aValid) {
      const unsigned short* src = aRow + k0;
      __builtin_prefetch((const void*)(src + 64), 0, 0);   // next K-slice
#ifdef USE_ASYNC_LDS
      __builtin_amdgcn_global_load_async_to_lds_b128(
          (gv4p)src, (lv4p)&As[amL][akk], 0, 0);
#else
      *(uint4*)&As[amL][akk] = *(const uint4*)src;
#endif
    } else {
      uint4 z = {0u, 0u, 0u, 0u};
      *(uint4*)&As[amL][akk] = z;
    }

    // ---- stage B: 128x64 im2col gather (incremental c/r/s walk) ----
    {
      int kg0 = k0 + bKK;
      int c = kg0 / khw;
      int rem = kg0 - c * khw;
      int r = rem / ks;
      int s = rem - r * ks;
      for (int j = 0; j < 32; ++j) {
        float v = 0.f;
        if (nValid && (kg0 + j) < Kg) {
          int hi = hib + r * dil;
          int wi = wib + s * dil;
          if (hi >= 0 && hi < H && wi >= 0 && wi < W)
            v = xb[(size_t)c * HW + hi * W + wi];
        }
        Bs[bRow][bKK + j] = f2bf(v);
        if (++s == ks) { s = 0; if (++r == ks) { r = 0; ++c; } }
      }
    }
#ifdef USE_ASYNC_LDS
    __builtin_amdgcn_s_wait_asynccnt(0);
#endif
    __syncthreads();

    // ---- 4 WMMAs: shared A fragment, two B fragments, K unrolled x2 ----
    const int half = lane >> 4;
    const int row  = lane & 15;
#pragma unroll
    for (int kt = 0; kt < 2; ++kt) {
      FragBF fa, fb0, fb1;
      fa.h[0]  = *(const ushort8*)&As[wm * 16 + row][kt * 32 + half * 8];
      fa.h[1]  = *(const ushort8*)&As[wm * 16 + row][kt * 32 + 16 + half * 8];
      fb0.h[0] = *(const ushort8*)&Bs[wn * 32 + row][kt * 32 + half * 8];
      fb0.h[1] = *(const ushort8*)&Bs[wn * 32 + row][kt * 32 + 16 + half * 8];
      fb1.h[0] = *(const ushort8*)&Bs[wn * 32 + 16 + row][kt * 32 + half * 8];
      fb1.h[1] = *(const ushort8*)&Bs[wn * 32 + 16 + row][kt * 32 + 16 + half * 8];
      acc[0] = __builtin_amdgcn_wmma_f32_16x16x32_bf16(
          false, fa.v, false, fb0.v, (short)0, acc[0], false, false);
      acc[1] = __builtin_amdgcn_wmma_f32_16x16x32_bf16(
          false, fa.v, false, fb1.v, (short)0, acc[1], false, false);
    }
    __syncthreads();
  }

  // ---- epilogue: folded BN + activation + residual, scatter to NCHW ----
  const int row  = lane & 15;
  const int half = lane >> 4;
  int nO[2], bO[2], hO[2], wO[2];
  bool nv[2];
#pragma unroll
  for (int t = 0; t < 2; ++t) {
    nO[t] = nBlk + wn * 32 + t * 16 + row;
    nv[t] = nO[t] < Ntot;
    int n = nv[t] ? nO[t] : 0;
    bO[t] = n / HoWo;
    int hw = n - bO[t] * HoWo;
    hO[t] = hw / Wo;
    wO[t] = hw - hO[t] * Wo;
  }
#pragma unroll
  for (int r = 0; r < 8; ++r) {
    int m = mBlk + wm * 16 + half * 8 + r;
    if (m < coutg) {
      int oc = gidx * coutg + m;
      float sc = 1.f, tb = 0.f;
      if (bn_g) {
        sc = bn_g[oc] * rsqrtf(bn_v[oc] + 1e-3f);
        tb = bn_b[oc] - bn_m[oc] * sc;
      }
#pragma unroll
      for (int t = 0; t < 2; ++t) {
        if (!nv[t]) continue;
        float vo = acc[t][r] * sc + tb;
        if (act == ACT_SILU)      vo = vo / (1.f + expf(-vo));
        else if (act == ACT_RELU) vo = fmaxf(vo, 0.f);
        size_t oi = (((size_t)bO[t] * Cout + oc) * Ho + hO[t]) * Wo + wO[t];
        if (res) vo += res[oi];
        y[oi] = vo;
      }
    }
  }
}

// ---------------------------- elementwise kernels --------------------------

__global__ void k_s2d(const float* __restrict__ x, float* __restrict__ y,
                      int B, int C, int H, int W) {
  int Ho = H / 2, Wo = W / 2;
  size_t n = (size_t)B * 4 * C * Ho * Wo;
  size_t i = (size_t)blockIdx.x * blockDim.x + threadIdx.x;
  if (i >= n) return;
  size_t t = i;
  int wo = t % Wo; t /= Wo;
  int ho = t % Ho; t /= Ho;
  int c4 = t % (4 * C); t /= 4 * C;
  int b = (int)t;
  int q = c4 / C, c = c4 % C;
  int dh = (q == 1 || q == 3) ? 1 : 0;
  int dw = (q >= 2) ? 1 : 0;
  y[i] = x[(((size_t)b * C + c) * H + 2 * ho + dh) * W + 2 * wo + dw];
}

// one block per (b,c) row; mean over HW of (x [+ x2]) with optional relu
__global__ void k_rowmean(const float* __restrict__ x, const float* __restrict__ x2,
                          float* __restrict__ out, int HW, int doRelu) {
  __shared__ float red[256];
  size_t row = blockIdx.x;
  const float* p  = x + row * (size_t)HW;
  const float* p2 = x2 ? x2 + row * (size_t)HW : nullptr;
  float s = 0.f;
  for (int i = threadIdx.x; i < HW; i += 256) {
    float v = p[i];
    if (p2) v += p2[i];
    if (doRelu) v = fmaxf(v, 0.f);
    s += v;
  }
  red[threadIdx.x] = s; __syncthreads();
  for (int o = 128; o > 0; o >>= 1) {
    if (threadIdx.x < o) red[threadIdx.x] += red[threadIdx.x + o];
    __syncthreads();
  }
  if (threadIdx.x == 0) out[row] = red[0] / (float)HW;
}

__global__ void k_div(const float* __restrict__ x, const float* __restrict__ m,
                      float* __restrict__ y, int HW, size_t n) {
  size_t i = (size_t)blockIdx.x * blockDim.x + threadIdx.x;
  if (i < n) y[i] = x[i] / (m[i / HW] + 1e-4f);
}

__global__ void k_gelu(float* __restrict__ y, size_t n) {
  size_t i = (size_t)blockIdx.x * blockDim.x + threadIdx.x;
  if (i < n) { float v = y[i]; y[i] = 0.5f * v * (1.f + erff(v * 0.70710678f)); }
}

__global__ void k_mul(const float* __restrict__ a, const float* __restrict__ b,
                      float* __restrict__ y, size_t n) {
  size_t i = (size_t)blockIdx.x * blockDim.x + threadIdx.x;
  if (i < n) y[i] = a[i] * b[i];
}

__global__ void k_copych(float* __restrict__ dst, const float* __restrict__ src,
                         int Cs, int HW, int Cd, int coff, size_t n) {
  size_t i = (size_t)blockIdx.x * blockDim.x + threadIdx.x;
  if (i >= n) return;
  int hw = (int)(i % HW);
  size_t t = i / HW;
  int c = (int)(t % Cs);
  int b = (int)(t / Cs);
  dst[((size_t)b * Cd + coff + c) * HW + hw] = src[i];
}

__global__ void k_maxpool(const float* __restrict__ x, float* __restrict__ y,
                          int B, int C, int H, int W, int ks) {
  int pad = ks / 2;
  size_t n = (size_t)B * C * H * W;
  size_t i = (size_t)blockIdx.x * blockDim.x + threadIdx.x;
  if (i >= n) return;
  int w0 = (int)(i % W);
  size_t t = i / W;
  int h0 = (int)(t % H); t /= H;
  const float* base = x + t * (size_t)H * W;  // (b,c) plane
  float m = -3.402823e38f;
  for (int r = 0; r < ks; ++r) {
    int hi = h0 - pad + r;
    if (hi < 0 || hi >= H) continue;
    for (int s = 0; s < ks; ++s) {
      int wi = w0 - pad + s;
      if (wi < 0 || wi >= W) continue;
      m = fmaxf(m, base[(size_t)hi * W + wi]);
    }
  }
  y[i] = m;
}

__global__ void k_up2(const float* __restrict__ x, float* __restrict__ y,
                      int B, int C, int H, int W) {
  int Ho = H * 2, Wo = W * 2;
  size_t n = (size_t)B * C * Ho * Wo;
  size_t i = (size_t)blockIdx.x * blockDim.x + threadIdx.x;
  if (i >= n) return;
  int wo = (int)(i % Wo);
  size_t t = i / Wo;
  int ho = (int)(t % Ho); t /= Ho;
  y[i] = x[t * (size_t)H * W + (size_t)(ho >> 1) * W + (wo >> 1)];
}

__global__ void k_skcomb(const float* __restrict__ o0, const float* __restrict__ o1,
                         const float* __restrict__ ab, float* __restrict__ y,
                         int C, int HW, size_t n) {
  size_t i = (size_t)blockIdx.x * blockDim.x + threadIdx.x;
  if (i >= n) return;
  size_t bc = i / HW;
  int c = (int)(bc % C);
  int b = (int)(bc / C);
  float a0 = ab[(size_t)b * 2 * C + c];
  float a1 = ab[(size_t)b * 2 * C + C + c];
  float mx = fmaxf(a0, a1);
  float e0 = expf(a0 - mx), e1 = expf(a1 - mx);
  float w0 = e0 / (e0 + e1);
  y[i] = o0[i] * w0 + o1[i] * (1.f - w0);
}

// ------------------------------- host side ---------------------------------

namespace {

struct CB { const float *w, *g, *b, *m, *v; int cin, cout, ks, groups; };
struct CSP { CB c1, c2, c3; CB m1[3], m2[3]; int n; };
struct SAG { CB fc1, fc2, mix, rw1, rw2; };
struct SK  { CB b0, b1, fc1; const float* fc2w; int d, c; };

struct Cur {
  void* const* in; int i;
  const float* next() { return (const float*)in[i++]; }
};

CB takeCB(Cur& c, int cin, int cout, int ks, int groups = 1) {
  CB r; r.w = c.next(); r.g = c.next(); r.b = c.next(); r.m = c.next(); r.v = c.next();
  r.cin = cin; r.cout = cout; r.ks = ks; r.groups = groups; return r;
}
CSP takeCSP(Cur& c, int cin, int cout, int n) {
  CSP r; int hid = cout / 2;
  r.c1 = takeCB(c, cin, hid, 1); r.c2 = takeCB(c, cin, hid, 1);
  r.c3 = takeCB(c, 2 * hid, cout, 1); r.n = n;
  for (int i = 0; i < n; ++i) { r.m1[i] = takeCB(c, hid, hid, 1); r.m2[i] = takeCB(c, hid, hid, 3); }
  return r;
}
SAG takeSAG(Cur& c, int d) {
  SAG r; r.fc1 = takeCB(c, d, d, 1); r.fc2 = takeCB(c, d, d, 3);
  r.mix = takeCB(c, 2 * d, d, 3); r.rw1 = takeCB(c, d, d, 1); r.rw2 = takeCB(c, d, d, 1);
  return r;
}
SK takeSK(Cur& c, int cin, int cout) {
  SK r; r.c = cout; r.d = (cin / 16 > 32) ? cin / 16 : 32;
  r.b0.w = c.next(); r.b0.g = c.next(); r.b0.b = c.next(); r.b0.m = c.next(); r.b0.v = c.next();
  r.b0.cin = cin; r.b0.cout = cout; r.b0.ks = 3; r.b0.groups = 32;
  r.b1 = r.b0;
  r.b1.w = c.next(); r.b1.g = c.next(); r.b1.b = c.next(); r.b1.m = c.next(); r.b1.v = c.next();
  r.fc1 = takeCB(c, cout, r.d, 1);
  r.fc2w = c.next();
  return r;
}

struct Ten { float* p; int B, C, H, W; size_t n() const { return (size_t)B * C * H * W; } };

struct Arena {
  char* base; size_t off;
  float* alloc(size_t elems) { return (float*)allocb(elems * 4); }
  void* allocb(size_t bytes) {
    void* p = (void*)(base + off);
    off = (off + bytes + 255) & ~(size_t)255;
    return p;
  }
};

inline int nblk(size_t n) { return (int)((n + 255) / 256); }

void launch_conv(Arena& A, hipStream_t st, const CB& cb, Ten x, Ten y,
                 int stride, int dil, int act, const float* res) {
  int pad   = dil * (cb.ks - 1) / 2;
  int coutg = cb.cout / cb.groups;
  int Kg    = (cb.cin / cb.groups) * cb.ks * cb.ks;
  int Kp    = (Kg + 63) & ~63;
  int Ntot  = y.B * y.H * y.W;

  // pack weights -> bf16 padded; stream order makes arena reuse safe
  size_t mk = A.off;
  unsigned short* wp = (unsigned short*)A.allocb((size_t)cb.cout * Kp * 2);
  k_packw<<<nblk((size_t)cb.cout * Kp), 256, 0, st>>>(cb.w, wp, cb.cout, Kg, Kp);

  dim3 grid((Ntot + 127) / 128, (coutg + 31) / 32, cb.groups);
  conv_wmma<<<grid, 256, 0, st>>>(x.p, wp, cb.g, cb.b, cb.m, cb.v, res, y.p,
                                  x.B, cb.cin, x.H, x.W, cb.cout, y.H, y.W,
                                  cb.ks, stride, pad, dil, cb.groups, Kg, Kp, act);
  A.off = mk;
}

Ten conv_op(Arena& A, hipStream_t st, const CB& cb, Ten x,
            int stride = 1, int act = ACT_SILU, const float* res = nullptr, int dil = 1) {
  int Ho = (x.H - 1) / stride + 1;   // "same" padding with odd ks
  int Wo = (x.W - 1) / stride + 1;
  Ten y{A.alloc((size_t)x.B * cb.cout * Ho * Wo), x.B, cb.cout, Ho, Wo};
  launch_conv(A, st, cb, x, y, stride, dil, act, res);
  return y;
}

void copych(hipStream_t st, Ten dst, Ten src, int coff) {
  size_t n = src.n();
  k_copych<<<nblk(n), 256, 0, st>>>(dst.p, src.p, src.C, src.H * src.W, dst.C, coff, n);
}

Ten csp_op(Arena& A, hipStream_t st, const CSP& p, Ten x) {
  Ten out{A.alloc((size_t)x.B * p.c3.cout * x.H * x.W), x.B, p.c3.cout, x.H, x.W};
  size_t mk = A.off;
  Ten x1 = conv_op(A, st, p.c1, x);
  Ten x2 = conv_op(A, st, p.c2, x);
  for (int i = 0; i < p.n; ++i) {
    Ten t = conv_op(A, st, p.m1[i], x1);
    x1 = conv_op(A, st, p.m2[i], t, 1, ACT_SILU, x1.p);   // residual fused
  }
  int hid = x1.C;
  Ten cat{A.alloc((size_t)x.B * 2 * hid * x.H * x.W), x.B, 2 * hid, x.H, x.W};
  copych(st, cat, x1, 0);
  copych(st, cat, x2, hid);
  launch_conv(A, st, p.c3, cat, out, 1, 1, ACT_SILU, nullptr);
  A.off = mk;
  return out;
}

Ten sag_op(Arena& A, hipStream_t st, const SAG& p, Ten x) {
  Ten out{A.alloc(x.n()), x.B, x.C, x.H, x.W};
  size_t mk = A.off;
  float* mbuf = A.alloc((size_t)x.B * x.C);
  k_rowmean<<<x.B * x.C, 256, 0, st>>>(x.p, nullptr, mbuf, x.H * x.W, 1);
  Ten xd{A.alloc(x.n()), x.B, x.C, x.H, x.W};
  k_div<<<nblk(x.n()), 256, 0, st>>>(x.p, mbuf, xd.p, x.H * x.W, x.n());
  Ten x1 = conv_op(A, st, p.fc1, xd);
  Ten x2 = conv_op(A, st, p.fc2, x);
  Ten cat{A.alloc(2 * x.n()), x.B, 2 * x.C, x.H, x.W};
  copych(st, cat, x1, 0);
  copych(st, cat, x2, x.C);
  Ten xm = conv_op(A, st, p.mix, cat);
  Ten r1 = conv_op(A, st, p.rw1, xm);
  k_gelu<<<nblk(r1.n()), 256, 0, st>>>(r1.p, r1.n());
  Ten r2 = conv_op(A, st, p.rw2, r1);
  k_mul<<<nblk(x.n()), 256, 0, st>>>(x.p, r2.p, out.p, x.n());
  A.off = mk;
  return out;
}

Ten sk_op(Arena& A, hipStream_t st, const SK& p, Ten x, float* outp = nullptr) {
  Ten out{outp ? outp : A.alloc(x.n()), x.B, p.c, x.H, x.W};
  size_t mk = A.off;
  Ten o0 = conv_op(A, st, p.b0, x, 1, ACT_RELU, nullptr, 1);   // dil=1, pad=1
  Ten o1 = conv_op(A, st, p.b1, x, 1, ACT_RELU, nullptr, 2);   // dil=2, pad=2
  float* s = A.alloc((size_t)x.B * p.c);
  k_rowmean<<<x.B * p.c, 256, 0, st>>>(o0.p, o1.p, s, x.H * x.W, 0);   // mean(U)
  Ten st_{s, x.B, p.c, 1, 1};
  Ten z = conv_op(A, st, p.fc1, st_, 1, ACT_RELU);
  CB fc2{p.fc2w, nullptr, nullptr, nullptr, nullptr, p.d, 2 * p.c, 1, 1};
  Ten ab = conv_op(A, st, fc2, z, 1, ACT_NONE);                // (B,2C,1,1)
  k_skcomb<<<nblk(out.n()), 256, 0, st>>>(o0.p, o1.p, ab.p, out.p, p.c,
                                          x.H * x.W, out.n());
  A.off = mk;
  return out;
}

Ten spp_op(Arena& A, hipStream_t st, const CB& c1, const CB& c2, Ten x) {
  Ten out{A.alloc((size_t)x.B * c2.cout * x.H * x.W), x.B, c2.cout, x.H, x.W};
  size_t mk = A.off;
  Ten a = conv_op(A, st, c1, x);
  Ten cat{A.alloc(4 * a.n()), a.B, 4 * a.C, a.H, a.W};
  copych(st, cat, a, 0);
  Ten tmp{A.alloc(a.n()), a.B, a.C, a.H, a.W};
  int kss[3] = {5, 9, 13};
  for (int i = 0; i < 3; ++i) {
    k_maxpool<<<nblk(a.n()), 256, 0, st>>>(a.p, tmp.p, a.B, a.C, a.H, a.W, kss[i]);
    copych(st, cat, tmp, (i + 1) * a.C);
  }
  launch_conv(A, st, c2, cat, out, 1, 1, ACT_SILU, nullptr);
  A.off = mk;
  return out;
}

Ten up2_op(Arena& A, hipStream_t st, Ten x) {
  Ten y{A.alloc(4 * x.n()), x.B, x.C, 2 * x.H, 2 * x.W};
  k_up2<<<nblk(y.n()), 256, 0, st>>>(x.p, y.p, x.B, x.C, x.H, x.W);
  return y;
}

} // namespace

extern "C" void kernel_launch(void* const* d_in, const int* in_sizes, int n_in,
                              void* d_out, int out_size, void* d_ws, size_t ws_size,
                              hipStream_t stream) {
  (void)in_sizes; (void)n_in; (void)out_size; (void)ws_size;
  Cur cur{d_in, 0};
  const float* xin = cur.next();                      // (4,3,512,512)

  CB  stem_focus = takeCB(cur, 12, 32, 3);
  CB  stem_conv  = takeCB(cur, 32, 64, 3);
  CB  csp1_conv  = takeCB(cur, 64, 64, 3);
  CSP csp1       = takeCSP(cur, 64, 64, 1);
  CB  csp2_conv  = takeCB(cur, 64, 128, 3);
  CSP csp2       = takeCSP(cur, 128, 128, 3);
  SAG sag1a      = takeSAG(cur, 64);
  CB  sag1_conv  = takeCB(cur, 64, 128, 3);
  SAG sag1b      = takeSAG(cur, 128);
  CB  sag2_conv  = takeCB(cur, 128, 128, 3);
  SAG sag2       = takeSAG(cur, 128);
  CB  sk1_conv   = takeCB(cur, 256, 256, 3);
  SK  sk1a       = takeSK(cur, 256, 256);
  CSP sk1_csp    = takeCSP(cur, 256, 256, 3);
  SK  sk1b       = takeSK(cur, 256, 256);
  CB  sk2_conv   = takeCB(cur, 256, 512, 3);
  SK  sk2a       = takeSK(cur, 512, 512);
  CSP sk2_csp    = takeCSP(cur, 512, 512, 3);
  SK  sk2b       = takeSK(cur, 512, 512);
  CB  spp1       = takeCB(cur, 512, 256, 1);
  CB  spp2       = takeCB(cur, 1024, 512, 1);
  CB  up1_conv   = takeCB(cur, 512, 256, 3);
  CB  csp3_conv  = takeCB(cur, 512, 256, 3);
  CSP csp3       = takeCSP(cur, 256, 256, 1);
  CB  up2_conv   = takeCB(cur, 256, 128, 3);
  CB  sk3_conv   = takeCB(cur, 384, 256, 3);
  SK  sk3a       = takeSK(cur, 256, 256);
  CSP sk3_csp    = takeCSP(cur, 256, 128, 3);
  SK  sk3b       = takeSK(cur, 128, 128);

  Arena A{(char*)d_ws, 0};

  // stem: focus (space-to-depth + 3x3) then stride-2 conv
  Ten x0{(float*)xin, 4, 3, 512, 512};
  Ten s2d{A.alloc((size_t)4 * 12 * 256 * 256), 4, 12, 256, 256};
  k_s2d<<<nblk(s2d.n()), 256, 0, stream>>>(x0.p, s2d.p, 4, 3, 512, 512);
  Ten xs = conv_op(A, stream, stem_focus, s2d);           // (4,32,256,256)
  Ten x  = conv_op(A, stream, stem_conv, xs, 2);          // (4,64,128,128)

  // left branch
  Ten left = csp_op(A, stream, csp1, conv_op(A, stream, csp1_conv, x, 1));
  left     = csp_op(A, stream, csp2, conv_op(A, stream, csp2_conv, left, 2)); // (4,128,64,64)

  // right branch
  Ten right = sag_op(A, stream, sag1a, x);
  right     = sag_op(A, stream, sag1b, conv_op(A, stream, sag1_conv, right, 2));
  right     = sag_op(A, stream, sag2,  conv_op(A, stream, sag2_conv, right, 1)); // (4,128,64,64)

  // merge
  Ten cat1{A.alloc((size_t)4 * 256 * 64 * 64), 4, 256, 64, 64};
  copych(stream, cat1, left, 0);
  copych(stream, cat1, right, 128);

  Ten t = conv_op(A, stream, sk1_conv, cat1, 2);          // (4,256,32,32)
  t = sk_op(A, stream, sk1a, t);
  t = csp_op(A, stream, sk1_csp, t);
  Ten residual = sk_op(A, stream, sk1b, t);               // (4,256,32,32)

  t = conv_op(A, stream, sk2_conv, residual, 2);          // (4,512,16,16)
  t = sk_op(A, stream, sk2a, t);
  t = csp_op(A, stream, sk2_csp, t);
  t = sk_op(A, stream, sk2b, t);
  t = spp_op(A, stream, spp1, spp2, t);                   // (4,512,16,16)

  t = conv_op(A, stream, up1_conv, t, 1);                 // (4,256,16,16)
  Ten u = up2_op(A, stream, t);                           // (4,256,32,32)
  Ten cat2{A.alloc((size_t)4 * 512 * 32 * 32), 4, 512, 32, 32};
  copych(stream, cat2, u, 0);
  copych(stream, cat2, residual, 256);
  t = csp_op(A, stream, csp3, conv_op(A, stream, csp3_conv, cat2, 1)); // (4,256,32,32)

  t = conv_op(A, stream, up2_conv, t, 1);                 // (4,128,32,32)
  u = up2_op(A, stream, t);                               // (4,128,64,64)
  Ten cat3{A.alloc((size_t)4 * 384 * 64 * 64), 4, 384, 64, 64};
  copych(stream, cat3, u, 0);
  copych(stream, cat3, left, 128);
  copych(stream, cat3, right, 256);

  t = conv_op(A, stream, sk3_conv, cat3, 1);              // (4,256,64,64)
  t = sk_op(A, stream, sk3a, t);
  t = csp_op(A, stream, sk3_csp, t);                      // (4,128,64,64)
  sk_op(A, stream, sk3b, t, (float*)d_out);               // final -> d_out
}
